// AKConv_44023414784060
// MI455X (gfx1250) — matrix operand order repaired and checked by reading, not compiled, not run on hardware
//
#include <hip/hip_runtime.h>
#include <cstdint>

typedef _Float16 h16;
typedef _Float16 v16h __attribute__((ext_vector_type(16)));
typedef _Float16 v8h  __attribute__((ext_vector_type(8)));
typedef float    v8f  __attribute__((ext_vector_type(8)));

#define BATCH 4
#define C1    128
#define C2    256
#define HH    64
#define WW    64
#define HWPIX 4096   // 64*64
#define NTAP  9

// ---------------------------------------------------------------------------
// Kernel 0: pack weights into f16 layouts friendly to WMMA A-operand loads.
//   pwp[c2][c1]   (f16, contiguous in c1)
//   wp [n][o][c]  (f16, contiguous in c)  from dcn_w[o][c][n]
// ---------------------------------------------------------------------------
__global__ void pack_weights(const float* __restrict__ pw_w,
                             const float* __restrict__ dcn_w,
                             h16* __restrict__ pwp,
                             h16* __restrict__ wp) {
    int i = blockIdx.x * 256 + threadIdx.x;
    if (i < C2 * C1) pwp[i] = (h16)pw_w[i];
    if (i < NTAP * C2 * C2) {
        int n = i / (C2 * C2);
        int r = i % (C2 * C2);
        int o = r / C2;
        int c = r % C2;
        wp[i] = (h16)dcn_w[(o * C2 + c) * NTAP + n];
    }
}

// ---------------------------------------------------------------------------
// Kernel 1: pointwise conv (GEMM 256x4096x128 per batch) + folded BN.
// One 128-thread block (4 waves) per (batch, 16-pixel) tile; the f16 B tile
// is staged once in LDS and shared by all 16 M-tiles (4 per wave).
// Outputs xp (f32, for offset conv) and xph (f16, sampling source).
// ---------------------------------------------------------------------------
__global__ void pw_bn_gemm(const float* __restrict__ x,
                           const h16*  __restrict__ pwp,
                           const float* __restrict__ gamma,
                           const float* __restrict__ beta,
                           const float* __restrict__ mean,
                           const float* __restrict__ var,
                           float* __restrict__ xp,
                           h16*  __restrict__ xph) {
    __shared__ __align__(64) h16 Bt[16 * C1];   // Bt[pix][c1], c1 contiguous

    int b    = blockIdx.x / (HWPIX / 16);
    int pt   = blockIdx.x % (HWPIX / 16);
    int pix0 = pt * 16;
    int tid  = threadIdx.x;

    // ---- stage B tile: x[b][0..127][pix0..pix0+15] -> f16 Bt[pix][c] ----
    {
        int pix = tid & 15;
        int cs  = (tid >> 4) * 16;                 // 8 groups of 16 channels
        const float* xs = x + ((size_t)b * C1 + cs) * HWPIX + pix0 + pix;
        v8h lo, hi;
#pragma unroll
        for (int i = 0; i < 8; ++i) lo[i] = (h16)xs[(size_t)i * HWPIX];
#pragma unroll
        for (int i = 0; i < 8; ++i) hi[i] = (h16)xs[(size_t)(i + 8) * HWPIX];
        *(v8h*)(Bt + pix * C1 + cs)     = lo;
        *(v8h*)(Bt + pix * C1 + cs + 8) = hi;
    }
    __syncthreads();

    int wave = tid >> 5;
    int lane = tid & 31;
    int m    = lane & 15;
    int half = lane >> 4;
    int pixcol = lane & 15;

    v8f acc[4];
#pragma unroll
    for (int i = 0; i < 4; ++i) acc[i] = (v8f){};

    for (int c0 = 0; c0 < C1; c0 += 32) {
        // B operand: 16 contiguous halfs at [pixcol][c0 + 16*half]
        v16h Bv = *(const v16h*)(Bt + pixcol * C1 + c0 + 16 * half);
#pragma unroll
        for (int mt = 0; mt < 4; ++mt) {
            int o = (wave * 4 + mt) * 16 + m;
            const h16* ar = pwp + o * C1 + c0;
            v8h alo = *(const v8h*)(ar + 16 * half);
            v8h ahi = *(const v8h*)(ar + 16 * half + 8);
            v16h A;
#pragma unroll
            for (int j = 0; j < 8; ++j) { A[j] = alo[j]; A[j + 8] = ahi[j]; }
            acc[mt] = __builtin_amdgcn_wmma_f32_16x16x32_f16(
                false, A, false, Bv, (short)0, acc[mt], false, false);
        }
    }

    // ---- BN fold + dual store (f32 + f16) ----
#pragma unroll
    for (int mt = 0; mt < 4; ++mt) {
#pragma unroll
        for (int r = 0; r < 8; ++r) {
            int ch  = (wave * 4 + mt) * 16 + r + 8 * half;
            float s  = gamma[ch] * rsqrtf(var[ch] + 1e-5f);
            float bb = beta[ch] - mean[ch] * s;
            float v  = acc[mt][r] * s + bb;
            size_t oi = ((size_t)b * C2 + ch) * HWPIX + pix0 + pixcol;
            xp[oi]  = v;
            xph[oi] = (h16)v;
        }
    }
}

// ---------------------------------------------------------------------------
// Kernel 2: 3x3 offset conv, 256 -> 18 channels, pad=1, f32 direct.
// Tiny (1.4 GFLOP) and L2-resident; scalar to avoid wasting WMMA M-rows.
// ---------------------------------------------------------------------------
__global__ void offset_conv(const float* __restrict__ xp,
                            const float* __restrict__ off_w,
                            const float* __restrict__ off_b,
                            float* __restrict__ off) {
    int idx = blockIdx.x * blockDim.x + threadIdx.x;
    if (idx >= BATCH * 18 * HWPIX) return;
    int wq = idx & 63;
    int t  = idx >> 6;
    int hq = t & 63; t >>= 6;
    int oc = t % 18;
    int b  = t / 18;

    float acc = off_b[oc];
    const float* xb   = xp + (size_t)b * C2 * HWPIX;
    const float* wrow = off_w + (size_t)oc * C2 * 9;
    for (int c = 0; c < C2; ++c) {
        const float* xc = xb + (size_t)c * HWPIX;
        const float* wc = wrow + c * 9;
#pragma unroll
        for (int dy = 0; dy < 3; ++dy) {
            int yy = hq + dy - 1;
            if ((unsigned)yy >= HH) continue;
#pragma unroll
            for (int dx = 0; dx < 3; ++dx) {
                int xx = wq + dx - 1;
                if ((unsigned)xx >= WW) continue;
                acc = fmaf(xc[yy * WW + xx], wc[dy * 3 + dx], acc);
            }
        }
    }
    off[idx] = acc;
}

// ---------------------------------------------------------------------------
// Kernel 3: deformable bilinear sampling + einsum (K=2304 GEMM) + SiLU.
// Block = 128 threads (4 waves), one (batch, 16-pixel) tile per block.
// Per tap n: sample f16 xph into S[pix][c] (8KB LDS), WMMA-accumulate.
// ---------------------------------------------------------------------------
__global__ void deform_dcn(const h16*  __restrict__ xph,
                           const float* __restrict__ off,
                           const h16*  __restrict__ wp,
                           float* __restrict__ out) {
    __shared__ __align__(64) h16 S[16 * C2];   // S[pix][c], c contiguous

    int b    = blockIdx.x / (HWPIX / 16);
    int pt   = blockIdx.x % (HWPIX / 16);
    int pix0 = pt * 16;

    int tid  = threadIdx.x;
    int wave = tid >> 5;
    int lane = tid & 31;
    int m    = lane & 15;
    int half = lane >> 4;
    int pixcol = lane & 15;

    // sampling-phase mapping
    int spix = tid & 15;
    int cb   = tid >> 4;          // 0..7 -> 32 channels each
    int p    = pix0 + spix;
    int hq   = p >> 6;
    int wq   = p & 63;

    const h16* xb = xph + (size_t)b * C2 * HWPIX;

    v8f acc[4];
#pragma unroll
    for (int i = 0; i < 4; ++i) acc[i] = (v8f){};

    for (int n = 0; n < NTAP; ++n) {
        __syncthreads();   // previous GEMM finished reading S

        // pull this tap's weight panel toward the caches (global_prefetch_b8)
        __builtin_prefetch(wp + ((size_t)n * C2 + wave * 64) * C2, 0, 1);

        // ---- bilinear coordinates for this thread's pixel (f32) ----
        float oy = off[((size_t)(b * 18 + n)     * HWPIX) + p];
        float ox = off[((size_t)(b * 18 + 9 + n) * HWPIX) + p];
        float py = (float)hq + (float)(n / 3 - 1) + oy;
        float px = (float)wq + (float)(n % 3 - 1) + ox;
        py = fminf(fmaxf(py, 0.f), (float)(HH - 1));
        px = fminf(fmaxf(px, 0.f), (float)(WW - 1));
        float y0f = floorf(py), x0f = floorf(px);
        float fy = py - y0f, fx = px - x0f;
        int y0 = (int)y0f, x0 = (int)x0f;
        int y1 = min(y0 + 1, HH - 1), x1 = min(x0 + 1, WW - 1);
        int i00 = y0 * WW + x0, i01 = y0 * WW + x1;
        int i10 = y1 * WW + x0, i11 = y1 * WW + x1;
        float w00 = (1.f - fy) * (1.f - fx), w01 = (1.f - fy) * fx;
        float w10 = fy * (1.f - fx),         w11 = fy * fx;

        // ---- fill S[pix][c] for 32 channels (f16 taps, f32 interpolation) ----
        for (int i = 0; i < 32; ++i) {
            int c = cb * 32 + i;
            const h16* xc = xb + (size_t)c * HWPIX;
            float v = (float)xc[i00] * w00 + (float)xc[i01] * w01
                    + (float)xc[i10] * w10 + (float)xc[i11] * w11;
            S[spix * C2 + c] = (h16)v;
        }
        __syncthreads();   // S ready

        // ---- WMMA accumulate: out_tile += W_n[o, c] * S_n[c, pix] ----
        for (int c0 = 0; c0 < C2; c0 += 32) {
            v16h Bv = *(const v16h*)(S + pixcol * C2 + c0 + 16 * half);
#pragma unroll
            for (int mt = 0; mt < 4; ++mt) {
                const h16* ab = wp
                    + ((size_t)(n * C2 + (wave * 4 + mt) * 16 + m)) * C2 + c0;
                v8h alo = *(const v8h*)(ab + 16 * half);
                v8h ahi = *(const v8h*)(ab + 16 * half + 8);
                v16h A;
#pragma unroll
                for (int j = 0; j < 8; ++j) { A[j] = alo[j]; A[j + 8] = ahi[j]; }
                acc[mt] = __builtin_amdgcn_wmma_f32_16x16x32_f16(
                    false, A, false, Bv, (short)0, acc[mt], false, false);
            }
        }
    }

    // ---- SiLU + store ----
#pragma unroll
    for (int mt = 0; mt < 4; ++mt) {
#pragma unroll
        for (int r = 0; r < 8; ++r) {
            int o = (wave * 4 + mt) * 16 + r + 8 * half;
            float v = acc[mt][r];
            float s = v / (1.f + __expf(-v));
            out[((size_t)b * C2 + o) * HWPIX + pix0 + pixcol] = s;
        }
    }
}

// ---------------------------------------------------------------------------
extern "C" void kernel_launch(void* const* d_in, const int* in_sizes, int n_in,
                              void* d_out, int out_size, void* d_ws, size_t ws_size,
                              hipStream_t stream) {
    const float* x     = (const float*)d_in[0];
    const float* pw_w  = (const float*)d_in[1];
    const float* gamma = (const float*)d_in[2];
    const float* beta  = (const float*)d_in[3];
    const float* mean  = (const float*)d_in[4];
    const float* var   = (const float*)d_in[5];
    const float* off_w = (const float*)d_in[6];
    const float* off_b = (const float*)d_in[7];
    const float* dcn_w = (const float*)d_in[8];
    float* out = (float*)d_out;

    // workspace: xp f32 | xph f16 | off f32 | pwp f16 | wp f16  (~27.6 MB)
    char* ws = (char*)d_ws;
    float* xp  = (float*)ws;                       // 16,777,216 B
    h16*   xph = (h16*)  (ws + 16777216);          //  8,388,608 B
    float* off = (float*)(ws + 25165824);          //  1,179,648 B
    h16*   pwp = (h16*)  (ws + 26345472);          //     65,536 B
    h16*   wp  = (h16*)  (ws + 26411008);          //  1,179,648 B

    // 0) weight packing / f16 conversion
    pack_weights<<<(NTAP * C2 * C2 + 255) / 256, 256, 0, stream>>>(pw_w, dcn_w, pwp, wp);

    // 1) pointwise GEMM + BN: one block per (batch, 16-pixel) tile
    pw_bn_gemm<<<BATCH * (HWPIX / 16), 128, 0, stream>>>(x, pwp, gamma, beta, mean, var, xp, xph);

    // 2) offset conv: 4*18*4096 outputs
    offset_conv<<<(BATCH * 18 * HWPIX + 255) / 256, 256, 0, stream>>>(xp, off_w, off_b, off);

    // 3) deformable sampling + einsum + SiLU
    deform_dcn<<<BATCH * (HWPIX / 16), 128, 0, stream>>>(xph, off, wp, out);
}
